// AdaptiveRNN_64089501991323
// MI455X (gfx1250) — matrix operand order repaired and checked by reading, not compiled, not run on hardware
//
#include <hip/hip_runtime.h>

#define BD   8
#define TD   2048
#define ID   1024
#define HD   1024

typedef __attribute__((ext_vector_type(16))) __bf16 v16bf;
typedef __attribute__((ext_vector_type(8)))  __bf16 v8bf;
typedef __attribute__((ext_vector_type(8)))  float  v8f;

#define WMMA_BF16(A, B, C) \
  __builtin_amdgcn_wmma_f32_16x16x32_bf16(false, (A), false, (B), (short)0, (C), false, false)

__device__ __forceinline__ v16bf ld2x8bf(const __bf16* p0, const __bf16* p1) {
  union { v16bf v; v8bf h[2]; } u;
  u.h[0] = *(const v8bf*)p0;
  u.h[1] = *(const v8bf*)p1;
  return u.v;
}
__device__ __forceinline__ v16bf ld16bf(const __bf16* p) {
  return ld2x8bf(p, p + 8);
}

__device__ __forceinline__ void split_bf(float f, __bf16& hi, __bf16& lo) {
  hi = (__bf16)f;
  lo = (__bf16)(f - (float)hi);
}

// ---------------------------------------------------------------------------
// Kernel 0: split W_ih and W_hh (fp32, HxI row-major) into bf16 hi/lo pairs.
// ---------------------------------------------------------------------------
__global__ void convert_weights(const float* __restrict__ Wih,
                                const float* __restrict__ Whh,
                                __bf16* __restrict__ wih_hi, __bf16* __restrict__ wih_lo,
                                __bf16* __restrict__ whh_hi, __bf16* __restrict__ whh_lo) {
  size_t i = (size_t)blockIdx.x * blockDim.x + threadIdx.x;   // 0 .. 2^20-1
  if (i >= (size_t)HD * ID) return;
  __bf16 h, l;
  split_bf(Wih[i], h, l); wih_hi[i] = h; wih_lo[i] = l;
  split_bf(Whh[i], h, l); whh_hi[i] = h; whh_lo[i] = l;
}

// ---------------------------------------------------------------------------
// Kernel 1: init padded (16 x H) hidden-state ping/pong buffers + barrier state.
// Rows 0..7 = h0, rows 8..15 = 0 (never rewritten -> WMMA padding rows).
// ---------------------------------------------------------------------------
__global__ void init_state(const float* __restrict__ h0,
                           __bf16* __restrict__ hhi0, __bf16* __restrict__ hlo0,
                           __bf16* __restrict__ hhi1, __bf16* __restrict__ hlo1,
                           int* __restrict__ cnt, int* __restrict__ gen) {
  int i = blockIdx.x * blockDim.x + threadIdx.x;   // 0 .. 32767  (2 * 16 * 1024)
  if (i == 0) { *cnt = 0; *gen = 0; }
  int buf = i >> 14;
  int rem = i & 16383;
  int m = rem >> 10, n = rem & 1023;
  float v = (m < BD) ? h0[m * HD + n] : 0.0f;
  __bf16 h, l; split_bf(v, h, l);
  if (buf == 0) { hhi0[rem] = h; hlo0[rem] = l; }
  else          { hhi1[rem] = h; hlo1[rem] = l; }
}

// ---------------------------------------------------------------------------
// Kernel 2: xw = x @ W_ih^T + (b_ih+b_hh), bf16x3 WMMA, result -> d_out.
// Each wave: one 16-row M tile x 64-col N group (4 WMMA tiles), K=1024.
// ---------------------------------------------------------------------------
__global__ void __launch_bounds__(128) gemm_xw(
    const float* __restrict__ x,
    const __bf16* __restrict__ wih_hi, const __bf16* __restrict__ wih_lo,
    const float* __restrict__ bih, const float* __restrict__ bhh,
    float* __restrict__ out) {
  const int wave = blockIdx.x * (blockDim.x >> 5) + (threadIdx.x >> 5);
  const int lane = threadIdx.x & 31;
  const int mtile  = wave >> 4;          // 0..1023  (B*T/16 tiles)
  const int ngroup = wave & 15;          // 0..15    (64 cols each)
  const int l15 = lane & 15;
  const int hiHalf = lane >> 4;          // 0 or 1
  const int k0 = hiHalf * 8;             // A-frag K phase
  const int kb = hiHalf * 16;            // unused; B frag also uses k0
  const int nbase = ngroup * 64;
  const size_t arow = (size_t)(mtile * 16 + l15) * ID;

  v8f acc[4];
  #pragma unroll
  for (int j = 0; j < 4; ++j) acc[j] = (v8f)(0.0f);

  for (int kk = 0; kk < ID / 32; ++kk) {
    const int kbase = kk * 32;
    // ---- A fragment: 16 fp32 of x -> bf16 hi/lo split, on the fly ----
    const float4* pa0 = (const float4*)(x + arow + kbase + k0);
    const float4* pa1 = (const float4*)(x + arow + kbase + k0 + 16);
    float4 f0 = pa0[0], f1 = pa0[1], f2 = pa1[0], f3 = pa1[1];
    float tf[16] = { f0.x, f0.y, f0.z, f0.w, f1.x, f1.y, f1.z, f1.w,
                     f2.x, f2.y, f2.z, f2.w, f3.x, f3.y, f3.z, f3.w };
    v16bf Ah, Al;
    #pragma unroll
    for (int j = 0; j < 16; ++j) {
      __bf16 h, l; split_bf(tf[j], h, l);
      Ah[j] = h; Al[j] = l;
    }
    // ---- 4 N-subtiles, pre-split W_ih, 3 WMMAs each (hi*hi + hi*lo + lo*hi)
    #pragma unroll
    for (int j = 0; j < 4; ++j) {
      const size_t boff = (size_t)(nbase + j * 16 + l15) * ID + kbase + k0;
      v16bf Bh = ld2x8bf(wih_hi + boff, wih_hi + boff + 16);
      v16bf Bl = ld2x8bf(wih_lo + boff, wih_lo + boff + 16);
      acc[j] = WMMA_BF16(Ah, Bh, acc[j]);
      acc[j] = WMMA_BF16(Ah, Bl, acc[j]);
      acc[j] = WMMA_BF16(Al, Bh, acc[j]);
    }
  }
  // Epilogue: D layout -> lanes 0-15: M=r, lanes 16-31: M=8+r; N = lane&15.
  const int mloc = hiHalf * 8;
  #pragma unroll
  for (int j = 0; j < 4; ++j) {
    const int n = nbase + j * 16 + l15;
    const float bias = bih[n] + bhh[n];
    #pragma unroll
    for (int r = 0; r < 8; ++r) {
      const size_t row = (size_t)(mtile * 16 + mloc + r);
      out[row * HD + n] = acc[j][r] + bias;
    }
  }
}

// ---------------------------------------------------------------------------
// Grid barrier (sense-reversing, atomic counter + generation).
// ---------------------------------------------------------------------------
__device__ __forceinline__ void grid_barrier(int* cnt, int* gen, int nblk) {
  __syncthreads();
  if (threadIdx.x == 0) {
    int g = __atomic_load_n(gen, __ATOMIC_ACQUIRE);
    __threadfence();
    int prev = __atomic_fetch_add(cnt, 1, __ATOMIC_ACQ_REL);
    if (prev == nblk - 1) {
      __atomic_store_n(cnt, 0, __ATOMIC_RELAXED);
      __atomic_fetch_add(gen, 1, __ATOMIC_RELEASE);
    } else {
      while (__atomic_load_n(gen, __ATOMIC_ACQUIRE) == g)
        __builtin_amdgcn_s_sleep(1);
    }
  }
  __syncthreads();
}

// ---------------------------------------------------------------------------
// Kernel 3: persistent cooperative scan.
//   h_new = tanh(xw_t + h @ W_hh^T);  out = h_new + x;  ping-pong bf16 hi/lo h.
// 16 blocks x 128 threads = 64 waves = 64 N-tiles of 16 columns.
// Each WG caches its 64 W_hh columns (hi+lo bf16 = 256 KB) in LDS once; the
// 2048-step serial loop then feeds all B fragments from ds_load instead of L2.
// ---------------------------------------------------------------------------
__global__ void __launch_bounds__(128) rnn_scan(
    const float* __restrict__ x, float* __restrict__ out,
    const __bf16* __restrict__ whh_hi, const __bf16* __restrict__ whh_lo,
    __bf16* hhi0, __bf16* hlo0, __bf16* hhi1, __bf16* hlo1,
    int* cnt, int* gen, float* __restrict__ hn) {
  extern __shared__ __bf16 lds[];
  __bf16* shh = lds;                 // [64][HD] hi slice of W_hh for this WG
  __bf16* shl = lds + 64 * HD;       // [64][HD] lo slice

  const int wave = blockIdx.x * (blockDim.x >> 5) + (threadIdx.x >> 5);
  const int lane = threadIdx.x & 31;
  const int l15 = lane & 15;
  const int hiHalf = lane >> 4;
  const int k0 = hiHalf * 8;
  const int kb = hiHalf * 16;
  const int n = wave * 16 + l15;                  // global output column
  const int colLocal = (threadIdx.x >> 5) * 16 + l15;   // 0..63 within WG
  const int ma = l15;                             // A-fragment row (padded 16)
  const int nblk = gridDim.x;

  // ---- one-time LDS fill: this WG's 64 columns of W_hh hi/lo ----
  {
    const v8bf* gh = (const v8bf*)(whh_hi + (size_t)blockIdx.x * 64 * HD);
    const v8bf* gl = (const v8bf*)(whh_lo + (size_t)blockIdx.x * 64 * HD);
    v8bf* sh = (v8bf*)shh;
    v8bf* sl = (v8bf*)shl;
    for (int i = threadIdx.x; i < 64 * HD / 8; i += (int)blockDim.x) {
      sh[i] = gh[i];
      sl[i] = gl[i];
    }
  }
  __syncthreads();

  for (int t = 0; t < TD; ++t) {
    const __bf16* chi = (t & 1) ? hhi1 : hhi0;
    const __bf16* clo = (t & 1) ? hlo1 : hlo0;
    __bf16* nhi = (t & 1) ? hhi0 : hhi1;
    __bf16* nlo = (t & 1) ? hlo0 : hlo1;

    v8f a0 = (v8f)(0.0f), a1 = (v8f)(0.0f), a2 = (v8f)(0.0f);
    for (int kk = 0; kk < HD / 32; ++kk) {
      const int kbase = kk * 32;
      v16bf Ah = ld2x8bf(chi + ma * HD + kbase + k0, chi + ma * HD + kbase + k0 + 16);
      v16bf Al = ld2x8bf(clo + ma * HD + kbase + k0, clo + ma * HD + kbase + k0 + 16);
      // B fragments from LDS (ds_load_b128): W_hh columns cached above
      const int soff = colLocal * HD + kbase + k0;
      union { v16bf v; v8bf h[2]; } ubh, ubl;
      ubh.h[0] = *(const v8bf*)&shh[soff];
      ubh.h[1] = *(const v8bf*)&shh[soff + 16];
      ubl.h[0] = *(const v8bf*)&shl[soff];
      ubl.h[1] = *(const v8bf*)&shl[soff + 16];
      a0 = WMMA_BF16(Ah, ubh.v, a0);   // three independent accumulation chains
      a1 = WMMA_BF16(Ah, ubl.v, a1);
      a2 = WMMA_BF16(Al, ubh.v, a2);
    }
    v8f s = a0 + a1 + a2;

    if (hiHalf == 0) {             // lanes 0-15 hold the real batch rows M=0..7
      #pragma unroll
      for (int r = 0; r < 8; ++r) {
        const size_t idx = ((size_t)r * TD + t) * HD + n;
        const float hv = tanhf(s[r] + out[idx]);   // out currently holds xw
        out[idx] = hv + x[idx];                    // fused residual add
        __bf16 hh, hl; split_bf(hv, hh, hl);
        nhi[r * HD + n] = hh;
        nlo[r * HD + n] = hl;
        if (t == TD - 1) hn[(size_t)r * HD + n] = hv;
      }
    }
    grid_barrier(cnt, gen, nblk);
  }
}

// ---------------------------------------------------------------------------
extern "C" void kernel_launch(void* const* d_in, const int* in_sizes, int n_in,
                              void* d_out, int out_size, void* d_ws, size_t ws_size,
                              hipStream_t stream) {
  const float* x   = (const float*)d_in[0];
  const float* h0  = (const float*)d_in[1];
  const float* Wih = (const float*)d_in[2];
  const float* Whh = (const float*)d_in[3];
  const float* bih = (const float*)d_in[4];
  const float* bhh = (const float*)d_in[5];
  float* out = (float*)d_out;
  float* hn  = out + (size_t)BD * TD * HD;

  char* ws = (char*)d_ws;
  int* cnt = (int*)ws;
  int* gen = (int*)(ws + 64);
  const size_t HB = (size_t)16 * HD * sizeof(__bf16);       // 32 KB
  __bf16* hhi0 = (__bf16*)(ws + 256);
  __bf16* hhi1 = (__bf16*)(ws + 256 + HB);
  __bf16* hlo0 = (__bf16*)(ws + 256 + 2 * HB);
  __bf16* hlo1 = (__bf16*)(ws + 256 + 3 * HB);
  const size_t WB = (size_t)HD * ID * sizeof(__bf16);       // 2 MB
  char* wsw = ws + 256 + 4 * HB;
  __bf16* wih_hi = (__bf16*)(wsw);
  __bf16* wih_lo = (__bf16*)(wsw + WB);
  __bf16* whh_hi = (__bf16*)(wsw + 2 * WB);
  __bf16* whh_lo = (__bf16*)(wsw + 3 * WB);

  convert_weights<<<(HD * ID + 255) / 256, 256, 0, stream>>>(
      Wih, Whh, wih_hi, wih_lo, whh_hi, whh_lo);
  init_state<<<256, 128, 0, stream>>>(h0, hhi0, hlo0, hhi1, hlo1, cnt, gen);
  // 16384 waves: (B*T/16) M-tiles x 16 N-groups; 4 waves / block.
  gemm_xw<<<4096, 128, 0, stream>>>(x, wih_hi, wih_lo, bih, bhh, out);
  // persistent cooperative scan: 16 blocks x 4 waves = 64 column tiles.
  // 256 KB dynamic LDS per WG caches that WG's W_hh slice (hi+lo bf16).
  rnn_scan<<<16, 128, 256 * 1024, stream>>>(x, out, whh_hi, whh_lo,
                                            hhi0, hlo0, hhi1, hlo1, cnt, gen, hn);
}
